// SingleStreamBlock_4690104287432
// MI455X (gfx1250) — matrix operand order repaired and checked by reading, not compiled, not run on hardware
//
#include <hip/hip_runtime.h>
#include <math.h>

typedef __bf16 bf16;
typedef __attribute__((ext_vector_type(8)))  bf16  v8bf;
typedef __attribute__((ext_vector_type(16))) bf16  v16bf;
typedef __attribute__((ext_vector_type(8)))  float v8f;

#define LQ   4096
#define DIM  3072
#define NH   24
#define HD   128
#define N1   21504   // 3*DIM + MLP
#define NCAT 15360   // DIM + MLP

__device__ __forceinline__ v16bf cat16(v8bf a, v8bf b) {
  return __builtin_shufflevector(a, b, 0,1,2,3,4,5,6,7,8,9,10,11,12,13,14,15);
}
__device__ __forceinline__ v8f vzero8() {
  v8f z = {0.f,0.f,0.f,0.f,0.f,0.f,0.f,0.f};
  return z;
}
__device__ __forceinline__ float gelu_tanh(float v) {
  return 0.5f * v * (1.f + tanhf(0.7978845608028654f * (v + 0.044715f * v * v * v)));
}

// CDNA5 async copy: global -> LDS, 16 bytes per lane, tracked by ASYNCcnt.
// Low 32 bits of a generic pointer to __shared__ are the LDS byte offset
// (flat LDS aperture keeps the offset in addr[31:0]).
__device__ __forceinline__ void async_ld16(void* lds_ptr, const void* gptr) {
  unsigned l = (unsigned)(size_t)lds_ptr;
  unsigned long long g = (unsigned long long)(size_t)gptr;
  asm volatile("global_load_async_to_lds_b128 %0, %1, off"
               :: "v"(l), "v"(g) : "memory");
}
__device__ __forceinline__ void wait_async_le8() {
  asm volatile("s_wait_asynccnt 0x8" ::: "memory");
}
__device__ __forceinline__ void wait_async_0() {
  asm volatile("s_wait_asynccnt 0x0" ::: "memory");
}

// ---------------------------------------------------------------------------
// 1) silu(vec)
// ---------------------------------------------------------------------------
__global__ void k_silu(const float* __restrict__ vec, float* __restrict__ sv) {
  int i = blockIdx.x * 256 + threadIdx.x;
  if (i < DIM) {
    float v = vec[i];
    sv[i] = v / (1.f + __expf(-v));
  }
}

// ---------------------------------------------------------------------------
// 2) m = silu(vec) @ mod_w.T + mod_b  (one wave per output row)
// ---------------------------------------------------------------------------
__global__ __launch_bounds__(256) void k_mod_gemv(const float* __restrict__ sv,
                                                  const float* __restrict__ mw,
                                                  const float* __restrict__ mb,
                                                  float* __restrict__ m) {
  int j = blockIdx.x * 8 + (threadIdx.x >> 5);
  int lane = threadIdx.x & 31;
  const float* wr = mw + (size_t)j * DIM;
  float acc = 0.f;
  for (int i = lane; i < DIM; i += 32) acc += sv[i] * wr[i];
  #pragma unroll
  for (int mk = 16; mk >= 1; mk >>= 1) acc += __shfl_xor(acc, mk, 32);
  if (lane == 0) m[j] = acc + mb[j];
}

// ---------------------------------------------------------------------------
// 3) x_mod = (1+scale)*layer_norm(x) + shift -> bf16
// ---------------------------------------------------------------------------
__global__ __launch_bounds__(256) void k_ln_mod(const float* __restrict__ x,
                                                const float* __restrict__ m,
                                                bf16* __restrict__ xmod) {
  int l = blockIdx.x;
  const float* xr = x + (size_t)l * DIM;
  __shared__ float red0[8], red1[8];
  int tid = threadIdx.x, lane = tid & 31, wid = tid >> 5;
  float s = 0.f, s2 = 0.f;
  for (int i = tid; i < DIM; i += 256) { float v = xr[i]; s += v; s2 += v * v; }
  #pragma unroll
  for (int mk = 16; mk >= 1; mk >>= 1) { s += __shfl_xor(s, mk, 32); s2 += __shfl_xor(s2, mk, 32); }
  if (lane == 0) { red0[wid] = s; red1[wid] = s2; }
  __syncthreads();
  float ts = 0.f, ts2 = 0.f;
  #pragma unroll
  for (int w = 0; w < 8; ++w) { ts += red0[w]; ts2 += red1[w]; }
  float mu = ts * (1.f / DIM);
  float var = ts2 * (1.f / DIM) - mu * mu;
  float rr = rsqrtf(var + 1e-6f);
  for (int i = tid; i < DIM; i += 256) {
    float v = (xr[i] - mu) * rr;
    xmod[(size_t)l * DIM + i] = (bf16)((1.f + m[DIM + i]) * v + m[i]);
  }
}

// ---------------------------------------------------------------------------
// 4) f32 -> bf16 weight conversion
// ---------------------------------------------------------------------------
__global__ __launch_bounds__(256) void k_cvt_bf16(const float* __restrict__ src,
                                                  bf16* __restrict__ dst, size_t n) {
  size_t i = ((size_t)blockIdx.x * 256 + threadIdx.x) * 4;
  if (i + 3 < n) {
    float4 v = *(const float4*)&src[i];
    dst[i + 0] = (bf16)v.x; dst[i + 1] = (bf16)v.y;
    dst[i + 2] = (bf16)v.z; dst[i + 3] = (bf16)v.w;
  }
}

// ---------------------------------------------------------------------------
// 5) WMMA GEMM: C[M][N] = A[M][K] * B[N][K]^T + bias, fused epilogues.
//    128x128 block tile, 8 waves in 4x2 grid, per-wave 32x64 = 2x4 WMMA tiles.
//    K panels double-buffered in LDS via async-to-LDS copies (ASYNCcnt).
// ---------------------------------------------------------------------------
template <int EPI>
__global__ __launch_bounds__(256) void k_gemm_bf16(
    const bf16* __restrict__ A, int lda,
    const bf16* __restrict__ Bw, int ldb,
    int K,
    const float* __restrict__ bias,
    float* __restrict__ qkf, bf16* __restrict__ vT, bf16* __restrict__ cat, // EPI 1
    const float* __restrict__ xres, const float* __restrict__ gate,         // EPI 2
    float* __restrict__ outp) {
  __shared__ __align__(16) bf16 As[2][128][64];
  __shared__ __align__(16) bf16 Bs[2][128][64];
  const int tid = threadIdx.x;
  const int lane = tid & 31;
  const int wid = tid >> 5;
  const int half = lane >> 4;
  const int mn = lane & 15;
  const int waveM = wid >> 1;   // 0..3 (32 rows each)
  const int waveN = wid & 1;    // 0..1 (64 cols each)
  const int m0 = blockIdx.x * 128;
  const int n0 = blockIdx.y * 128;

  v8f acc[2][4];
  #pragma unroll
  for (int a = 0; a < 2; ++a)
    #pragma unroll
    for (int b = 0; b < 4; ++b) acc[a][b] = vzero8();

  // 8 async instructions per wave per staged tile.
  auto stage = [&](int buf, int kbase) {
    #pragma unroll
    for (int i = 0; i < 4; ++i) {
      int c = tid + i * 256;
      int r = c >> 3;
      int kc = (c & 7) * 8;
      async_ld16(&As[buf][r][kc], &A[(size_t)(m0 + r) * lda + kbase + kc]);
      async_ld16(&Bs[buf][r][kc], &Bw[(size_t)(n0 + r) * ldb + kbase + kc]);
    }
  };

  const int ktiles = K >> 6;
  stage(0, 0);
  for (int kt = 0; kt < ktiles; ++kt) {
    const int buf = kt & 1;
    if (kt + 1 < ktiles) {
      stage(buf ^ 1, (kt + 1) << 6);
      wait_async_le8();   // next tile in flight; current tile landed
    } else {
      wait_async_0();
    }
    __syncthreads();
    #pragma unroll
    for (int kk = 0; kk < 64; kk += 32) {
      v16bf af[2];
      #pragma unroll
      for (int ms = 0; ms < 2; ++ms) {
        const bf16* pa = &As[buf][waveM * 32 + ms * 16 + mn][kk + half * 8];
        af[ms] = cat16(*(const v8bf*)pa, *(const v8bf*)(pa + 16));
      }
      #pragma unroll
      for (int ns = 0; ns < 4; ++ns) {
        const bf16* pb = &Bs[buf][waveN * 64 + ns * 16 + mn][kk + half * 8];
        v16bf bfr = cat16(*(const v8bf*)pb, *(const v8bf*)(pb + 16));
        #pragma unroll
        for (int ms = 0; ms < 2; ++ms)
          acc[ms][ns] = __builtin_amdgcn_wmma_f32_16x16x32_bf16(
              false, af[ms], false, bfr, (short)0, acc[ms][ns], false, false);
      }
    }
    __syncthreads();
  }

  // Epilogue. C layout: element i of v8f -> row (i + 8*half), col (lane&15).
  #pragma unroll
  for (int ms = 0; ms < 2; ++ms)
    #pragma unroll
    for (int ns = 0; ns < 4; ++ns)
      #pragma unroll
      for (int i = 0; i < 8; ++i) {
        int gm = m0 + waveM * 32 + ms * 16 + i + 8 * half;
        int gn = n0 + waveN * 64 + ns * 16 + mn;
        float v = acc[ms][ns][i] + bias[gn];
        if (EPI == 1) {
          if (gn < 3 * DIM) {
            int t = gn / DIM;           // 0=q 1=k 2=v
            int h = (gn % DIM) >> 7;
            int d = gn & 127;
            if (t < 2)
              qkf[(((size_t)t * NH + h) * LQ + gm) * HD + d] = v;
            else
              vT[((size_t)h * HD + d) * LQ + gm] = (bf16)v;   // V transposed
          } else {
            cat[(size_t)gm * NCAT + (gn - 3 * DIM) + DIM] = (bf16)gelu_tanh(v);
          }
        } else {
          outp[(size_t)gm * DIM + gn] =
              xres[(size_t)gm * DIM + gn] + gate[gn] * v;
        }
      }
}

// ---------------------------------------------------------------------------
// 6) RMSNorm + RoPE for q,k -> bf16 per-head [h][l][d]. One wave per (h,l).
// ---------------------------------------------------------------------------
__global__ __launch_bounds__(256) void k_qkprep(const float* __restrict__ qkf,
                                                const float* __restrict__ pe,
                                                const float* __restrict__ qsc,
                                                const float* __restrict__ ksc,
                                                bf16* __restrict__ qb,
                                                bf16* __restrict__ kb) {
  int idx = blockIdx.x * 8 + (threadIdx.x >> 5);   // 0..NH*LQ-1
  int h = idx >> 12;
  int l = idx & (LQ - 1);
  int lane = threadIdx.x & 31;
  int d = lane * 4;
  int p = d >> 1;
  float c0 = pe[((size_t)l * 64 + p) * 4 + 0];
  float s0 = pe[((size_t)l * 64 + p) * 4 + 2];
  float c1 = pe[((size_t)l * 64 + p + 1) * 4 + 0];
  float s1 = pe[((size_t)l * 64 + p + 1) * 4 + 2];
  #pragma unroll
  for (int t = 0; t < 2; ++t) {
    const float* src = qkf + (((size_t)t * NH + h) * LQ + l) * HD;
    float4 v = *(const float4*)&src[d];
    float ss = v.x * v.x + v.y * v.y + v.z * v.z + v.w * v.w;
    #pragma unroll
    for (int mk = 16; mk >= 1; mk >>= 1) ss += __shfl_xor(ss, mk, 32);
    float rr = rsqrtf(ss * (1.f / HD) + 1e-6f);
    const float* scl = (t == 0) ? qsc : ksc;
    float a0 = v.x * rr * scl[d + 0], a1 = v.y * rr * scl[d + 1];
    float a2 = v.z * rr * scl[d + 2], a3 = v.w * rr * scl[d + 3];
    bf16* dst = ((t == 0) ? qb : kb) + ((size_t)h * LQ + l) * HD + d;
    dst[0] = (bf16)(c0 * a0 - s0 * a1);
    dst[1] = (bf16)(s0 * a0 + c0 * a1);
    dst[2] = (bf16)(c1 * a2 - s1 * a3);
    dst[3] = (bf16)(s1 * a2 + c1 * a3);
  }
}

// ---------------------------------------------------------------------------
// 7) Flash attention with WMMA. Block = (q-tile of 128, head). 8 waves, each
//    owns 16 query rows. 64-key tiles streamed through LDS via async copies.
// ---------------------------------------------------------------------------
__global__ __launch_bounds__(256) void k_attn(const bf16* __restrict__ qb,
                                              const bf16* __restrict__ kb,
                                              const bf16* __restrict__ vT,
                                              bf16* __restrict__ cat) {
  const int h = blockIdx.y;
  const int q0 = blockIdx.x * 128;
  __shared__ __align__(16) bf16 Qs[128][128];
  __shared__ __align__(16) bf16 Ks[64][128];
  __shared__ __align__(16) bf16 Vt[128][64];   // [d][key]: P@V B-frags K-contiguous
  __shared__ __align__(16) bf16 Ps[128][64];
  const int tid = threadIdx.x;
  const int lane = tid & 31;
  const int wid = tid >> 5;
  const int half = lane >> 4;
  const int mn = lane & 15;

  const bf16* qbase = qb + ((size_t)h * LQ + q0) * HD;
  #pragma unroll
  for (int i = 0; i < 8; ++i) {
    int c = tid + i * 256;
    int r = c >> 4;
    int kc = (c & 15) * 8;
    async_ld16(&Qs[r][kc], &qbase[(size_t)r * HD + kc]);
  }

  float mrow[8], lrow[8];
  v8f o[8];
  #pragma unroll
  for (int i = 0; i < 8; ++i) { mrow[i] = -1e30f; lrow[i] = 0.f; o[i] = vzero8(); }

  const float sm = 0.08838834764831845f;  // HD^-0.5

  for (int kt = 0; kt < LQ / 64; ++kt) {
    const int k0 = kt * 64;
    const bf16* kbase = kb + ((size_t)h * LQ + k0) * HD;
    const bf16* vbase = vT + (size_t)h * HD * LQ + k0;
    #pragma unroll
    for (int i = 0; i < 4; ++i) {
      int c = tid + i * 256;
      int r = c >> 4;
      int kc = (c & 15) * 8;
      async_ld16(&Ks[r][kc], &kbase[(size_t)r * HD + kc]);
      int rv = c >> 3;
      int kv = (c & 7) * 8;
      async_ld16(&Vt[rv][kv], &vbase[(size_t)rv * LQ + kv]);
    }
    if (kt + 1 < LQ / 64) {  // streaming hint for the next key tile
      __builtin_prefetch(&kbase[(size_t)64 * HD + (tid >> 1)], 0, 1);
    }
    wait_async_0();
    __syncthreads();

    // S = Q @ K^T (per wave: 16 x 64, K = HD = 128)
    v8f s[4];
    #pragma unroll
    for (int ns = 0; ns < 4; ++ns) s[ns] = vzero8();
    #pragma unroll
    for (int kk = 0; kk < HD; kk += 32) {
      const bf16* pa = &Qs[wid * 16 + mn][kk + half * 8];
      v16bf af = cat16(*(const v8bf*)pa, *(const v8bf*)(pa + 16));
      #pragma unroll
      for (int ns = 0; ns < 4; ++ns) {
        const bf16* pb = &Ks[ns * 16 + mn][kk + half * 8];
        v16bf bfr = cat16(*(const v8bf*)pb, *(const v8bf*)(pb + 16));
        s[ns] = __builtin_amdgcn_wmma_f32_16x16x32_bf16(
            false, af, false, bfr, (short)0, s[ns], false, false);
      }
    }
    // online softmax (rows live in 16-lane halves: xor masks 1,2,4,8)
    float mcur[8];
    #pragma unroll
    for (int i = 0; i < 8; ++i) {
      float mx = s[0][i] * sm;
      #pragma unroll
      for (int ns = 1; ns < 4; ++ns) mx = fmaxf(mx, s[ns][i] * sm);
      mcur[i] = mx;
    }
    #pragma unroll
    for (int mk = 1; mk < 16; mk <<= 1)
      #pragma unroll
      for (int i = 0; i < 8; ++i) mcur[i] = fmaxf(mcur[i], __shfl_xor(mcur[i], mk, 32));
    float psum[8];
    #pragma unroll
    for (int i = 0; i < 8; ++i) {
      float mnew = fmaxf(mrow[i], mcur[i]);
      float f = __expf(mrow[i] - mnew);
      mrow[i] = mnew;
      lrow[i] *= f;
      #pragma unroll
      for (int ns = 0; ns < 8; ++ns) o[ns][i] *= f;
      psum[i] = 0.f;
      #pragma unroll
      for (int ns = 0; ns < 4; ++ns) {
        float pv = __expf(s[ns][i] * sm - mnew);
        s[ns][i] = pv;
        psum[i] += pv;
      }
    }
    #pragma unroll
    for (int mk = 1; mk < 16; mk <<= 1)
      #pragma unroll
      for (int i = 0; i < 8; ++i) psum[i] += __shfl_xor(psum[i], mk, 32);
    #pragma unroll
    for (int i = 0; i < 8; ++i) lrow[i] += psum[i];

    // stage P in LDS (per-wave private rows), re-enter as A fragments
    #pragma unroll
    for (int ns = 0; ns < 4; ++ns)
      #pragma unroll
      for (int i = 0; i < 8; ++i)
        Ps[wid * 16 + i + 8 * half][ns * 16 + mn] = (bf16)s[ns][i];

    // O += P @ V   (per wave: 16 x 128, K = 64 keys)
    #pragma unroll
    for (int kk = 0; kk < 64; kk += 32) {
      const bf16* pa = &Ps[wid * 16 + mn][kk + half * 8];
      v16bf af = cat16(*(const v8bf*)pa, *(const v8bf*)(pa + 16));
      #pragma unroll
      for (int ns = 0; ns < 8; ++ns) {
        const bf16* pb = &Vt[ns * 16 + mn][kk + half * 8];
        v16bf bfr = cat16(*(const v8bf*)pb, *(const v8bf*)(pb + 16));
        o[ns] = __builtin_amdgcn_wmma_f32_16x16x32_bf16(
            false, af, false, bfr, (short)0, o[ns], false, false);
      }
    }
    __syncthreads();
  }

  #pragma unroll
  for (int ns = 0; ns < 8; ++ns)
    #pragma unroll
    for (int i = 0; i < 8; ++i) {
      int gq = q0 + wid * 16 + i + 8 * half;
      int gd = ns * 16 + mn;
      cat[(size_t)gq * NCAT + h * HD + gd] = (bf16)(o[ns][i] / lrow[i]);
    }
}

// ---------------------------------------------------------------------------
// Host-side orchestration
// ---------------------------------------------------------------------------
extern "C" void kernel_launch(void* const* d_in, const int* in_sizes, int n_in,
                              void* d_out, int out_size, void* d_ws, size_t ws_size,
                              hipStream_t stream) {
  const float* x   = (const float*)d_in[0];
  const float* vec = (const float*)d_in[1];
  const float* pe  = (const float*)d_in[2];
  const float* w1  = (const float*)d_in[3];
  const float* b1  = (const float*)d_in[4];
  const float* w2  = (const float*)d_in[5];
  const float* b2  = (const float*)d_in[6];
  const float* mw  = (const float*)d_in[7];
  const float* mb  = (const float*)d_in[8];
  const float* qsc = (const float*)d_in[9];
  const float* ksc = (const float*)d_in[10];
  float* out = (float*)d_out;

  char* ws = (char*)d_ws;
  size_t off = 0;
  auto alloc = [&](size_t bytes) -> void* {
    void* p = ws + off;
    off = (off + bytes + 255) & ~(size_t)255;
    return p;
  };
  float* sv   = (float*)alloc((size_t)DIM * 4);
  float* m    = (float*)alloc((size_t)3 * DIM * 4);
  bf16* xmod  = (bf16*)alloc((size_t)LQ * DIM * 2);
  bf16* w1b   = (bf16*)alloc((size_t)N1 * DIM * 2);
  bf16* w2b   = (bf16*)alloc((size_t)DIM * NCAT * 2);
  float* qkf  = (float*)alloc((size_t)2 * NH * LQ * HD * 4);
  bf16* vT    = (bf16*)alloc((size_t)NH * HD * LQ * 2);
  bf16* qb    = (bf16*)alloc((size_t)NH * LQ * HD * 2);
  bf16* kb    = (bf16*)alloc((size_t)NH * LQ * HD * 2);
  bf16* cat   = (bf16*)alloc((size_t)LQ * NCAT * 2);

  k_silu<<<(DIM + 255) / 256, 256, 0, stream>>>(vec, sv);
  k_mod_gemv<<<(3 * DIM) / 8, 256, 0, stream>>>(sv, mw, mb, m);
  k_ln_mod<<<LQ, 256, 0, stream>>>(x, m, xmod);

  size_t n_w1 = (size_t)N1 * DIM;
  size_t n_w2 = (size_t)DIM * NCAT;
  k_cvt_bf16<<<(unsigned)(n_w1 / 1024), 256, 0, stream>>>(w1, w1b, n_w1);
  k_cvt_bf16<<<(unsigned)(n_w2 / 1024), 256, 0, stream>>>(w2, w2b, n_w2);

  // GEMM1: [4096 x 3072] x [21504 x 3072]^T
  k_gemm_bf16<1><<<dim3(LQ / 128, N1 / 128), 256, 0, stream>>>(
      xmod, DIM, w1b, DIM, DIM, b1, qkf, vT, cat, nullptr, nullptr, nullptr);

  k_qkprep<<<(NH * LQ) / 8, 256, 0, stream>>>(qkf, pe, qsc, ksc, qb, kb);

  k_attn<<<dim3(LQ / 128, NH), 256, 0, stream>>>(qb, kb, vT, cat);

  // GEMM2: [4096 x 15360] x [3072 x 15360]^T, residual epilogue
  k_gemm_bf16<2><<<dim3(LQ / 128, DIM / 128), 256, 0, stream>>>(
      cat, NCAT, w2b, NCAT, NCAT, b2, nullptr, nullptr, nullptr,
      x, m + 2 * DIM, out);
}
